// BloomAttention_76227079570131
// MI455X (gfx1250) — matrix operand order repaired and checked by reading, not compiled, not run on hardware
//
#include <hip/hip_runtime.h>

// ---------------------------------------------------------------------------
// BLOOM fused attention for MI455X (gfx1250), wave32, bf16 WMMA 16x16x32.
// B=2 H=16 Q=KV=2048 D=128, fp32 in/out, f32 accumulate, online softmax.
// K/V staged in LDS as packed bf16 pairs in FRAGMENT-MAJOR order so every
// WMMA B-fragment is exactly two ds_load_b128 into contiguous VGPR quads.
// Fragment loads are ping-pong double-buffered so WMMAs wait on dscnt<=2
// (overlap LDS latency with the matrix pipe) instead of dscnt==0.
// Double-buffered tiles, one barrier per KV tile, 8 waves per block.
// ---------------------------------------------------------------------------

typedef __attribute__((ext_vector_type(16))) __bf16 v16bf;
typedef __attribute__((ext_vector_type(8)))  float  v8f;

#define B_   2
#define H_   16
#define Qn   2048
#define KVn  2048
#define Dn   128
#define TILE_KV 32
#define WAVES 8
#define ROWS_PER_WAVE 16
#define Q_PER_BLOCK (WAVES * ROWS_PER_WAVE)   // 128 query rows per block
#define NT (KVn / TILE_KV)                    // 64 KV tiles

union FragU {
    v16bf v;
    unsigned int u[8];
    uint4 q[2];
};

__device__ inline unsigned int pack2_bf16(float lo, float hi) {
    union { __bf16 h[2]; unsigned int u; } x;
    x.h[0] = (__bf16)lo;            // -> v_cvt_pk_bf16_f32
    x.h[1] = (__bf16)hi;
    return x.u;
}

__device__ inline unsigned short f32_to_bf16_bits(float f) {
    unsigned int u = __builtin_bit_cast(unsigned int, f);
    u += 0x7fffu + ((u >> 16) & 1u);  // round-to-nearest-even
    return (unsigned short)(u >> 16);
}

// Stage one 32-wide KV tile of K and V into LDS, packed bf16 pairs in
// fragment-major order:
//   K word: W = (pk>>3)*256  + col*8 + (pk&7)   pk = d-pair,  col = kv (0..31)
//   V word: W = (pk>>3)*1024 + col*8 + (pk&7)   pk = kv-pair, col = d  (0..127)
// so a lane's 8-word B fragment is contiguous (two b128 loads).
__device__ inline void stage_tile(const float* __restrict__ Kg,  // + bh base
                                  const float* __restrict__ Vg,  // + bh base
                                  unsigned int* __restrict__ lKp,
                                  unsigned int* __restrict__ lVp,
                                  int tid, int kv0) {
    {   // K: 64 d-pairs x 32 kv = 2048 words; thread: pair p, 8 kv columns
        int p    = tid >> 2;              // 0..63
        int kvl0 = (tid & 3) * 8;         // 0,8,16,24
        const float* r0 = Kg + (size_t)(2 * p    ) * KVn + kv0 + kvl0;
        const float* r1 = Kg + (size_t)(2 * p + 1) * KVn + kv0 + kvl0;
        float4 a0 = *reinterpret_cast<const float4*>(r0);
        float4 a1 = *reinterpret_cast<const float4*>(r0 + 4);
        float4 b0 = *reinterpret_cast<const float4*>(r1);
        float4 b1 = *reinterpret_cast<const float4*>(r1 + 4);
        float ka[8] = {a0.x, a0.y, a0.z, a0.w, a1.x, a1.y, a1.z, a1.w};
        float kb[8] = {b0.x, b0.y, b0.z, b0.w, b1.x, b1.y, b1.z, b1.w};
        unsigned int base = (unsigned)(p >> 3) * 256u + (unsigned)(p & 7);
        #pragma unroll
        for (int c2 = 0; c2 < 8; ++c2)
            lKp[base + (unsigned)(kvl0 + c2) * 8u] = pack2_bf16(ka[c2], kb[c2]);
    }
    {   // V: 16 kv-pairs x 128 d = 2048 words; thread: pair q, 8 d columns
        int qq  = tid >> 4;               // 0..15
        int ds0 = (tid & 15) * 8;         // 0..120
        const float* r0 = Vg + (size_t)(kv0 + 2 * qq) * Dn + ds0;
        const float* r1 = r0 + Dn;
        float4 a0 = *reinterpret_cast<const float4*>(r0);
        float4 a1 = *reinterpret_cast<const float4*>(r0 + 4);
        float4 b0 = *reinterpret_cast<const float4*>(r1);
        float4 b1 = *reinterpret_cast<const float4*>(r1 + 4);
        float va[8] = {a0.x, a0.y, a0.z, a0.w, a1.x, a1.y, a1.z, a1.w};
        float vb[8] = {b0.x, b0.y, b0.z, b0.w, b1.x, b1.y, b1.z, b1.w};
        unsigned int base = (unsigned)(qq >> 3) * 1024u + (unsigned)(qq & 7);
        #pragma unroll
        for (int c2 = 0; c2 < 8; ++c2)
            lVp[base + (unsigned)(ds0 + c2) * 8u] = pack2_bf16(va[c2], vb[c2]);
    }
}

__global__ __launch_bounds__(256)
void bloom_attn_fwd(const float* __restrict__ Qp,
                    const float* __restrict__ Kp,   // [B,H,D,KV] pre-transposed
                    const float* __restrict__ Vp,   // [B,H,KV,D]
                    const float* __restrict__ Ap,   // [B*H,Q,KV] alibi
                    const float* __restrict__ Mp,   // [B,1,Q,KV] mask
                    float* __restrict__ Op)         // [B,Q,H*D]
{
    // packed bf16 tiles, double buffered: 16KB + 16KB, P staging 8KB = 40KB
    __shared__ __attribute__((aligned(16))) unsigned int lKp[2][(Dn / 2) * TILE_KV];
    __shared__ __attribute__((aligned(16))) unsigned int lVp[2][(TILE_KV / 2) * Dn];
    __shared__ __attribute__((aligned(16))) unsigned short lP[WAVES][ROWS_PER_WAVE * TILE_KV];

    const int tid  = threadIdx.x;
    const int wid  = tid >> 5;
    const int lane = tid & 31;
    const int nl   = lane & 15;       // N (col) index inside 16-wide tile
    const int hi   = lane >> 4;       // lane half per ISA A/B/C layouts
    const int hi8  = hi * 8;
    const int bh   = blockIdx.x;      // 0..31
    const int b    = bh >> 4;
    const int h    = bh & 15;
    const int q0   = blockIdx.y * Q_PER_BLOCK + wid * ROWS_PER_WAVE;

    const float inv_norm = 0.08838834764831845f;  // 1/sqrt(128)
    const float L2E      = 1.4426950408889634f;

    const float* Kg = Kp + (size_t)bh * Dn  * (size_t)KVn;
    const float* Vg = Vp + (size_t)bh * (size_t)KVn * Dn;
    const size_t abase = (size_t)bh * (size_t)Qn * KVn;
    const size_t mbase = (size_t)b  * (size_t)Qn * KVn;

    // ---- Q fragments (A-layout, prescaled by 1/sqrt(D), bf16), kept in regs
    FragU qf[4];
    {
        const float* qrow = Qp + ((size_t)bh * Qn + (size_t)(q0 + nl)) * Dn;
        #pragma unroll
        for (int c = 0; c < 4; ++c) {
            #pragma unroll
            for (int i = 0; i < 8; ++i) {
                // A 16x32 bf16, k order used here (K/V packing matches): VGPR i holds K pair 2i; lane-half adds +16
                int k0 = 2 * i + hi * 16;
                qf[c].u[i] = pack2_bf16(qrow[c * 32 + k0] * inv_norm,
                                        qrow[c * 32 + k0 + 1] * inv_norm);
            }
        }
    }

    // ---- output accumulator: 16x128 f32 in WMMA C layout (8 blocks of v8f)
    v8f oacc[8];
    #pragma unroll
    for (int j = 0; j < 8; ++j)
        #pragma unroll
        for (int e = 0; e < 8; ++e) oacc[j][e] = 0.0f;

    float m_run[8], l_run[8];
    #pragma unroll
    for (int r = 0; r < 8; ++r) { m_run[r] = -INFINITY; l_run[r] = 0.0f; }

    // prime the pipeline: stage tile 0 into buffer 0
    stage_tile(Kg, Vg, lKp[0], lVp[0], tid, 0);

    for (int kt = 0; kt < NT; ++kt) {
        const int kv0 = kt * TILE_KV;
        const int bi  = kt & 1;
        const unsigned int* lK = lKp[bi];
        const unsigned int* lV = lVp[bi];

        __syncthreads();   // staged tile kt visible; buffer bi^1 free to refill

        // issue next tile's global loads early (overlap with WMMA below)
        if (kt + 1 < NT) {
            stage_tile(Kg, Vg, lKp[bi ^ 1], lVp[bi ^ 1], tid, kv0 + TILE_KV);
            // prefetch next alibi tile rows for this wave (global_prefetch_b8)
            __builtin_prefetch(Ap + abase + (size_t)(q0 + nl) * KVn + kv0 + TILE_KV, 0, 0);
        }

        // ---- S = (Q/sqrtD) @ K : two independent 16x16 accumulator chains,
        // interleaved t0/t1, with ping-pong B-fragment prefetch.
        v8f sC[2];
        #pragma unroll
        for (int t = 0; t < 2; ++t)
            #pragma unroll
            for (int e = 0; e < 8; ++e) sC[t][e] = 0.0f;

        FragU fb[2];
        {
            const uint4* fp = reinterpret_cast<const uint4*>(
                &lK[(unsigned)(hi * TILE_KV + nl) * 8u]);     // (c=0,t=0)
            fb[0].q[0] = fp[0];
            fb[0].q[1] = fp[1];
        }
        #pragma unroll
        for (int step = 0; step < 8; ++step) {
            const int t = step & 1;
            const int c = step >> 1;
            if (step < 7) {   // prefetch next fragment into the other buffer
                const int tn = (step + 1) & 1;
                const int cn = (step + 1) >> 1;
                const uint4* fp = reinterpret_cast<const uint4*>(
                    &lK[(unsigned)((cn * 2 + hi) * TILE_KV + tn * 16 + nl) * 8u]);
                fb[(step + 1) & 1].q[0] = fp[0];
                fb[(step + 1) & 1].q[1] = fp[1];
            }
            sC[t] = __builtin_amdgcn_wmma_f32_16x16x32_bf16(
                false, qf[c].v, false, fb[step & 1].v, (short)0, sC[t],
                false, false);
        }

        // ---- scores = S + alibi + mask (C layout: VGPR r -> row r / r+8)
        float sc[2][8];
        #pragma unroll
        for (int t = 0; t < 2; ++t) {
            #pragma unroll
            for (int r = 0; r < 8; ++r) {
                int    row = q0 + r + hi8;
                size_t col = (size_t)kv0 + t * 16 + nl;
                float  al  = __builtin_nontemporal_load(
                                 Ap + abase + (size_t)row * KVn + col);
                float  mk  = Mp[mbase + (size_t)row * KVn + col];
                sc[t][r] = sC[t][r] + al + mk;
            }
        }

        // ---- online softmax + write P (bf16) to per-wave LDS row-major
        unsigned short* lPw = lP[wid];
        #pragma unroll
        for (int r = 0; r < 8; ++r) {
            float mx = fmaxf(sc[0][r], sc[1][r]);
            #pragma unroll
            for (int s = 8; s >= 1; s >>= 1)
                mx = fmaxf(mx, __shfl_xor(mx, s, 32));  // within 16-lane half
            float mnew  = fmaxf(m_run[r], mx);
            float alpha = exp2f((m_run[r] - mnew) * L2E);
            float p0    = exp2f((sc[0][r] - mnew) * L2E);
            float p1    = exp2f((sc[1][r] - mnew) * L2E);
            float rs    = p0 + p1;
            #pragma unroll
            for (int s = 8; s >= 1; s >>= 1)
                rs += __shfl_xor(rs, s, 32);
            l_run[r] = l_run[r] * alpha + rs;
            m_run[r] = mnew;
            #pragma unroll
            for (int j = 0; j < 8; ++j) oacc[j][r] *= alpha;
            int row = r + hi8;
            lPw[row * TILE_KV +  0 + nl] = f32_to_bf16_bits(p0);
            lPw[row * TILE_KV + 16 + nl] = f32_to_bf16_bits(p1);
        }
        asm volatile("s_wait_dscnt 0x0" ::: "memory");   // wave-local ordering

        // ---- gather P as A fragment: two contiguous uint4 runs in lP
        FragU pa;
        {
            const unsigned int* lPw32 = reinterpret_cast<const unsigned int*>(lPw);
            pa.q[0] = *reinterpret_cast<const uint4*>(&lPw32[nl * 16 + hi * 8]);
            pa.q[1] = *reinterpret_cast<const uint4*>(&lPw32[nl * 16 + hi * 8 + 4]);
        }

        // ---- O += P @ V : 8 independent accumulators, ping-pong B prefetch
        FragU vb2[2];
        {
            const uint4* fp = reinterpret_cast<const uint4*>(
                &lV[(unsigned)(hi * Dn + nl) * 8u]);          // j=0
            vb2[0].q[0] = fp[0];
            vb2[0].q[1] = fp[1];
        }
        #pragma unroll
        for (int j = 0; j < 8; ++j) {
            if (j < 7) {  // prefetch next fragment into the other buffer
                const uint4* fp = reinterpret_cast<const uint4*>(
                    &lV[(unsigned)(hi * Dn + (j + 1) * 16 + nl) * 8u]);
                vb2[(j + 1) & 1].q[0] = fp[0];
                vb2[(j + 1) & 1].q[1] = fp[1];
            }
            oacc[j] = __builtin_amdgcn_wmma_f32_16x16x32_bf16(
                false, pa.v, false, vb2[j & 1].v, (short)0, oacc[j],
                false, false);
        }
    }

    // ---- normalize and store: out[b, q, h*128 + d]
    const size_t ostride = (size_t)H_ * Dn;
    #pragma unroll
    for (int r = 0; r < 8; ++r) {
        int   row   = q0 + r + hi8;
        float inv_l = 1.0f / l_run[r];
        #pragma unroll
        for (int j = 0; j < 8; ++j) {
            Op[((size_t)b * Qn + row) * ostride + (size_t)h * Dn + j * 16 + nl] =
                oacc[j][r] * inv_l;
        }
    }
}

extern "C" void kernel_launch(void* const* d_in, const int* in_sizes, int n_in,
                              void* d_out, int out_size, void* d_ws, size_t ws_size,
                              hipStream_t stream) {
    (void)in_sizes; (void)n_in; (void)d_ws; (void)ws_size; (void)out_size;
    const float* Qp = (const float*)d_in[0];
    const float* Kp = (const float*)d_in[1];
    const float* Vp = (const float*)d_in[2];
    const float* Ap = (const float*)d_in[3];
    const float* Mp = (const float*)d_in[4];
    float* Op = (float*)d_out;

    dim3 grid(B_ * H_, Qn / Q_PER_BLOCK);   // (32, 16)
    dim3 block(256);                        // 8 waves (wave32)
    hipLaunchKernelGGL(bloom_attn_fwd, grid, block, 0, stream,
                       Qp, Kp, Vp, Ap, Mp, Op);
}
